// AIJNet_58798102282574
// MI455X (gfx1250) — matrix operand order repaired and checked by reading, not compiled, not run on hardware
//
#include <hip/hip_runtime.h>
#include <hip/hip_bf16.h>
#include <math.h>

typedef __bf16 bf16_t;
typedef __attribute__((ext_vector_type(16))) __bf16 v16bf;
typedef __attribute__((ext_vector_type(8)))  __bf16 v8bf;
typedef __attribute__((ext_vector_type(8)))  float  v8f;

#define TILE_M 128
#define TILE_N 128
#define TILE_K 32
#define LDS_PITCH 40   // 32 + 8 pad; all fragment offsets stay multiples of 8 elems (16B)

// Use CDNA5 async global->LDS copies (ASYNCcnt) when the toolchain exposes them.
#if defined(__has_builtin)
#  if __has_builtin(__builtin_amdgcn_global_load_async_to_lds_b128) && \
      __has_builtin(__builtin_amdgcn_s_wait_asynccnt)
#    define USE_ASYNC_LDS 1
#  endif
#endif
#ifndef USE_ASYNC_LDS
#  define USE_ASYNC_LDS 0
#endif

// exact pointee type the builtin expects: GCC-style int vector of 4
typedef __attribute__((__vector_size__(4 * sizeof(int)))) int v4i_t;
typedef __attribute__((address_space(1))) v4i_t as1_v4i;
typedef __attribute__((address_space(3))) v4i_t as3_v4i;

// ---------------------------------------------------------------------------
// bf16 WMMA GEMM, NT only:  Out = act( alpha * A @ Bt^T + bias )
//  A  : [M x Kd] bf16 row-major (pitch Kd), optional batch stride
//  Bt : [N x Kd] bf16 row-major (pitch Kd)  == op(B), optional batch stride
//  Out: bf16 or fp32 [M x N]
// 256 threads (8 waves), block tile 128x128x32, wave tile 32x64 (2x4 WMMA),
// double-buffered LDS, one barrier per k-step, async global->LDS staging.
// ---------------------------------------------------------------------------
template<bool RELU, bool HAS_BIAS, bool OUT_F32>
__global__ __launch_bounds__(256)
void gemm_wmma_nt(const bf16_t* __restrict__ A,  long long aBatch,
                  const bf16_t* __restrict__ Bt, long long bBatch,
                  const float*  __restrict__ bias,
                  void* __restrict__ OutV, long long oBatch,
                  int N, int Kd, float alpha)
{
    __shared__ bf16_t Alds[2][TILE_M * LDS_PITCH];
    __shared__ bf16_t Blds[2][TILE_N * LDS_PITCH];

    const int tid  = threadIdx.x;
    const int lane = tid & 31;
    const int wave = tid >> 5;
    const int wrow = (wave >> 1) * 32;   // 4 wave-rows
    const int wcol = (wave & 1) * 64;    // 2 wave-cols
    const int row0 = blockIdx.y * TILE_M;
    const int col0 = blockIdx.x * TILE_N;

    const bf16_t* Ab = A  + (long long)blockIdx.z * aBatch;
    const bf16_t* Bb = Bt + (long long)blockIdx.z * bBatch;

    v8f acc[2][4];
#pragma unroll
    for (int i = 0; i < 2; ++i)
#pragma unroll
        for (int j = 0; j < 4; ++j)
#pragma unroll
            for (int g = 0; g < 8; ++g) acc[i][j][g] = 0.0f;

    // staging decomposition: 4 threads per row (8 bf16 = 16B each), 2 passes
    const int sR = tid >> 2;           // 0..63
    const int sK = (tid & 3) * 8;      // 0,8,16,24

    // fragment addressing (ISA 16-bit WMMA layouts)
    const int fr  = lane & 15;
    const int akb = (lane < 16) ? 0 : 8;   // A: lanes16-31 hold K=8..15,24..31
    const int bkb = (lane < 16) ? 0 : 8;   // B: lanes16-31 hold K=8..15,24..31

    const int nK = Kd / TILE_K;

    auto compute = [&](int buf) {
        v16bf afrag[2];
#pragma unroll
        for (int i = 0; i < 2; ++i) {
            const bf16_t* base = &Alds[buf][(wrow + 16 * i + fr) * LDS_PITCH + akb];
            v8bf lo = *(const v8bf*)(base);        // K = akb .. akb+7
            v8bf hi = *(const v8bf*)(base + 16);   // K = akb+16 .. akb+23
            v16bf f;
#pragma unroll
            for (int e = 0; e < 8; ++e) { f[e] = lo[e]; f[e + 8] = hi[e]; }
            afrag[i] = f;
        }
        v16bf bfrag[4];
#pragma unroll
        for (int j = 0; j < 4; ++j) {
            const bf16_t* base = &Blds[buf][(wcol + 16 * j + fr) * LDS_PITCH + bkb];
            v8bf lo = *(const v8bf*)(base);        // K = bkb .. bkb+7
            v8bf hi = *(const v8bf*)(base + 16);   // K = bkb+16 .. bkb+23
            v16bf f;
#pragma unroll
            for (int e = 0; e < 8; ++e) { f[e] = lo[e]; f[e + 8] = hi[e]; }
            bfrag[j] = f;
        }
#pragma unroll
        for (int i = 0; i < 2; ++i)
#pragma unroll
            for (int j = 0; j < 4; ++j)
                acc[i][j] = __builtin_amdgcn_wmma_f32_16x16x32_bf16(
                    false, afrag[i], false, bfrag[j],
                    (short)0, acc[i][j], false, false);
    };

#if USE_ASYNC_LDS
    auto stage_async = [&](int buf, int k0) {
#pragma unroll
        for (int p = 0; p < 2; ++p) {
            const int r = sR + p * 64;
            v4i_t* ga = (v4i_t*)(Ab + (long long)(row0 + r) * Kd + k0 + sK);
            v4i_t* gb = (v4i_t*)(Bb + (long long)(col0 + r) * Kd + k0 + sK);
            v4i_t* la = (v4i_t*)&Alds[buf][r * LDS_PITCH + sK];
            v4i_t* lb = (v4i_t*)&Blds[buf][r * LDS_PITCH + sK];
            __builtin_amdgcn_global_load_async_to_lds_b128(
                (as1_v4i*)ga, (as3_v4i*)la, 0, 0);
            __builtin_amdgcn_global_load_async_to_lds_b128(
                (as1_v4i*)gb, (as3_v4i*)lb, 0, 0);
        }
    };
    stage_async(0, 0);
    __builtin_amdgcn_s_wait_asynccnt(0);
    __syncthreads();
    for (int kt = 0; kt < nK; ++kt) {
        const int cur = kt & 1;
        if (kt + 1 < nK) stage_async(cur ^ 1, (kt + 1) * TILE_K);  // overlaps compute
        compute(cur);
        __builtin_amdgcn_s_wait_asynccnt(0);
        __syncthreads();
    }
#else
    v8bf pa[2], pb[2];
    auto prefetch = [&](int k0) {
#pragma unroll
        for (int p = 0; p < 2; ++p) {
            const int r = sR + p * 64;
            pa[p] = *(const v8bf*)(Ab + (long long)(row0 + r) * Kd + k0 + sK);
            pb[p] = *(const v8bf*)(Bb + (long long)(col0 + r) * Kd + k0 + sK);
        }
    };
    auto commit = [&](int buf) {
#pragma unroll
        for (int p = 0; p < 2; ++p) {
            const int r = sR + p * 64;
            *(v8bf*)&Alds[buf][r * LDS_PITCH + sK] = pa[p];
            *(v8bf*)&Blds[buf][r * LDS_PITCH + sK] = pb[p];
        }
    };
    prefetch(0);
    commit(0);
    __syncthreads();
    for (int kt = 0; kt < nK; ++kt) {
        if (kt + 1 < nK) prefetch((kt + 1) * TILE_K);   // global loads in flight
        compute(kt & 1);                                 // overlap with loads
        if (kt + 1 < nK) commit((kt + 1) & 1);
        __syncthreads();
    }
#endif

    // ---- epilogue: C/D layout lanes0-15 M=g, lanes16-31 M=g+8, N=lane&15 ----
    const int cn   = lane & 15;
    const int rofs = (lane < 16) ? 0 : 8;
    const long long obase = (long long)blockIdx.z * oBatch;
#pragma unroll
    for (int i = 0; i < 2; ++i) {
#pragma unroll
        for (int j = 0; j < 4; ++j) {
            const int col = col0 + wcol + 16 * j + cn;
            float bv = HAS_BIAS ? bias[col] : 0.0f;
#pragma unroll
            for (int g = 0; g < 8; ++g) {
                const int row = row0 + wrow + 16 * i + rofs + g;
                float v = acc[i][j][g] * alpha + bv;
                if (RELU) v = fmaxf(v, 0.0f);
                if constexpr (OUT_F32)
                    ((float*)OutV)[obase + (long long)row * N + col] = v;
                else
                    ((bf16_t*)OutV)[obase + (long long)row * N + col] = (bf16_t)v;
            }
        }
    }
}

// ---------------------------------------------------------------------------
// X = bf16( concat(emb1, emb2, axis=-1) )    [rows x 2E]
// ---------------------------------------------------------------------------
__global__ __launch_bounds__(256)
void pack_concat_bf16(const float* __restrict__ e1, const float* __restrict__ e2,
                      bf16_t* __restrict__ X, int E)
{
    const long long idx = (long long)blockIdx.x * 256 + threadIdx.x;
    const int twoE = 2 * E;
    const long long r = idx / twoE;
    const int c = (int)(idx - r * twoE);
    float v = (c < E) ? e1[r * E + c] : e2[r * E + (c - E)];
    X[idx] = (bf16_t)v;
}

// ---------------------------------------------------------------------------
// dst[c][r] = bf16(src[r][c])  -- fused cast + transpose for fp32 weights
// 32x32 LDS tile; coalesced on both sides.
// ---------------------------------------------------------------------------
__global__ __launch_bounds__(256)
void cast_transpose_f32_bf16(const float* __restrict__ src, bf16_t* __restrict__ dst,
                             int R, int C)
{
    __shared__ float t[32][33];
    const int tx = threadIdx.x & 31;
    const int ty = threadIdx.x >> 5;       // 0..7
    const int c0 = blockIdx.x * 32;
    const int r0 = blockIdx.y * 32;
#pragma unroll
    for (int i = 0; i < 4; ++i)
        t[ty + i * 8][tx] = src[(long long)(r0 + ty + i * 8) * C + c0 + tx];
    __syncthreads();
#pragma unroll
    for (int i = 0; i < 4; ++i)
        dst[(long long)(c0 + ty + i * 8) * R + r0 + tx] = (bf16_t)t[tx][ty + i * 8];
}

// batched bf16 transpose: dst[b][c][r] = src[b][r][c]
__global__ __launch_bounds__(256)
void transpose_bf16_batched(const bf16_t* __restrict__ src, bf16_t* __restrict__ dst,
                            int R, int C, long long sB, long long dB)
{
    __shared__ bf16_t t[32][33];
    const int tx = threadIdx.x & 31;
    const int ty = threadIdx.x >> 5;
    const int c0 = blockIdx.x * 32;
    const int r0 = blockIdx.y * 32;
    const bf16_t* s = src + (long long)blockIdx.z * sB;
    bf16_t* d = dst + (long long)blockIdx.z * dB;
#pragma unroll
    for (int i = 0; i < 4; ++i)
        t[ty + i * 8][tx] = s[(long long)(r0 + ty + i * 8) * C + c0 + tx];
    __syncthreads();
#pragma unroll
    for (int i = 0; i < 4; ++i)
        d[(long long)(c0 + ty + i * 8) * R + r0 + tx] = t[tx][ty + i * 8];
}

// ---------------------------------------------------------------------------
// Row softmax with query-row mask (mask==0 -> -inf row, NaN like reference).
// ---------------------------------------------------------------------------
__global__ __launch_bounds__(256)
void softmax_mask(const float* __restrict__ scores,
                  const int* __restrict__ mask1, const int* __restrict__ mask2,
                  bf16_t* __restrict__ attn, int S, int Shalf)
{
    __shared__ float red[256];
    const int tid = threadIdx.x;
    const int row = blockIdx.x;
    const int b = row / S;
    const int q = row - b * S;
    const int m = (q < Shalf) ? mask1[b * Shalf + q] : mask2[b * Shalf + (q - Shalf)];

    const float* src = scores + (long long)row * S;
    float vals[4];
    float mx = -INFINITY;
#pragma unroll
    for (int i = 0; i < 4; ++i) {
        float v = src[tid + 256 * i];
        if (m == 0) v = -INFINITY;
        vals[i] = v;
        mx = fmaxf(mx, v);
    }
    red[tid] = mx; __syncthreads();
    for (int s = 128; s > 0; s >>= 1) {
        if (tid < s) red[tid] = fmaxf(red[tid], red[tid + s]);
        __syncthreads();
    }
    mx = red[0]; __syncthreads();

    float e[4]; float sum = 0.0f;
#pragma unroll
    for (int i = 0; i < 4; ++i) { e[i] = __expf(vals[i] - mx); sum += e[i]; }
    red[tid] = sum; __syncthreads();
    for (int s = 128; s > 0; s >>= 1) {
        if (tid < s) red[tid] += red[tid + s];
        __syncthreads();
    }
    const float inv = 1.0f / red[0];

    bf16_t* dst = attn + (long long)row * S;
#pragma unroll
    for (int i = 0; i < 4; ++i) dst[tid + 256 * i] = (bf16_t)(e[i] * inv);
}

// ---------------------------------------------------------------------------
// out[row] = sigmoid( dot(h2[row,:], W3) + b3 )
// ---------------------------------------------------------------------------
__global__ __launch_bounds__(256)
void head_sigmoid(const bf16_t* __restrict__ h2, const float* __restrict__ W3,
                  const float* __restrict__ b3, float* __restrict__ out, int Kd)
{
    __shared__ float red[256];
    const int tid = threadIdx.x;
    const long long row = blockIdx.x;
    const bf16_t* src = h2 + row * Kd;
    float s = 0.0f;
    for (int k = tid; k < Kd; k += 256) s += (float)src[k] * W3[k];
    red[tid] = s; __syncthreads();
    for (int t = 128; t > 0; t >>= 1) {
        if (tid < t) red[tid] += red[tid + t];
        __syncthreads();
    }
    if (tid == 0) {
        float z = red[0] + b3[0];
        out[row] = 1.0f / (1.0f + __expf(-z));
    }
}

// ---------------------------------------------------------------------------
extern "C" void kernel_launch(void* const* d_in, const int* in_sizes, int n_in,
                              void* d_out, int out_size, void* d_ws, size_t ws_size,
                              hipStream_t stream) {
    (void)in_sizes; (void)n_in; (void)out_size; (void)ws_size;

    constexpr int Bn = 16, S = 1024, E = 512, D = 1024, H1 = 2048, H2 = 1024;
    constexpr int M = Bn * S;                  // 16384 rows
    const float SCALE = 0.044194173824159216f; // 1/sqrt(512)

    const float* emb1 = (const float*)d_in[0];
    const float* emb2 = (const float*)d_in[1];
    const int*   mask1 = (const int*)d_in[2];
    const int*   mask2 = (const int*)d_in[3];
    const float* Wq = (const float*)d_in[4];  const float* bq = (const float*)d_in[5];
    const float* Wk = (const float*)d_in[6];  const float* bk = (const float*)d_in[7];
    const float* Wv = (const float*)d_in[8];  const float* bv = (const float*)d_in[9];
    const float* W1 = (const float*)d_in[10]; const float* b1 = (const float*)d_in[11];
    const float* W2 = (const float*)d_in[12]; const float* b2 = (const float*)d_in[13];
    const float* W3 = (const float*)d_in[14]; const float* b3 = (const float*)d_in[15];

    // ---- workspace layout (all 256B aligned) ----
    unsigned char* ws = (unsigned char*)d_ws;
    size_t off = 0;
    auto alloc = [&](size_t bytes) { size_t o = off; off += (bytes + 255) & ~(size_t)255; return o; };
    bf16_t* X    = (bf16_t*)(ws + alloc((size_t)M * D * 2));       // reused for `attended`
    bf16_t* Wqt  = (bf16_t*)(ws + alloc((size_t)D * D * 2));       // [D_out x D_in]
    bf16_t* Wkt  = (bf16_t*)(ws + alloc((size_t)D * D * 2));
    bf16_t* Wvt  = (bf16_t*)(ws + alloc((size_t)D * D * 2));
    bf16_t* W1t  = (bf16_t*)(ws + alloc((size_t)H1 * D * 2));      // [H1 x D]
    bf16_t* W2t  = (bf16_t*)(ws + alloc((size_t)H2 * H1 * 2));     // [H2 x H1]
    bf16_t* Qb   = (bf16_t*)(ws + alloc((size_t)M * D * 2));       // Q+K region reused for h1
    bf16_t* Kb   = (bf16_t*)(ws + alloc((size_t)M * D * 2));
    bf16_t* Vb   = (bf16_t*)(ws + alloc((size_t)M * D * 2));       // reused for h2
    bf16_t* Vt   = (bf16_t*)(ws + alloc((size_t)M * D * 2));       // [B][D][S]
    float*  Sc   = (float*) (ws + alloc((size_t)Bn * S * S * 4));
    bf16_t* At   = (bf16_t*)(ws + alloc((size_t)Bn * S * S * 2));
    bf16_t* Attd = X;    // [M x D] bf16, X no longer needed after QKV
    bf16_t* h1   = Qb;   // [M x H1] bf16 = 64MB -> exactly Q+K region
    bf16_t* h2   = Vb;   // [M x H2] bf16 -> V region (free after V transpose)

    // 0) pack concat(emb1,emb2) -> bf16 X; cast+transpose weights -> bf16 [N x K]
    pack_concat_bf16<<<(M * D) / 256, 256, 0, stream>>>(emb1, emb2, X, E);
    cast_transpose_f32_bf16<<<dim3(D / 32, D / 32), 256, 0, stream>>>(Wq, Wqt, D, D);
    cast_transpose_f32_bf16<<<dim3(D / 32, D / 32), 256, 0, stream>>>(Wk, Wkt, D, D);
    cast_transpose_f32_bf16<<<dim3(D / 32, D / 32), 256, 0, stream>>>(Wv, Wvt, D, D);
    cast_transpose_f32_bf16<<<dim3(H1 / 32, D / 32), 256, 0, stream>>>(W1, W1t, D, H1);
    cast_transpose_f32_bf16<<<dim3(H2 / 32, H1 / 32), 256, 0, stream>>>(W2, W2t, H1, H2);

    // 1) QKV projections: [M x D] = X @ W + b   (NT vs W^T, bias, bf16 out)
    {
        dim3 g(D / TILE_N, M / TILE_M, 1);
        gemm_wmma_nt<false, true, false><<<g, 256, 0, stream>>>(
            X, 0, Wqt, 0, bq, Qb, 0, D, D, 1.0f);
        gemm_wmma_nt<false, true, false><<<g, 256, 0, stream>>>(
            X, 0, Wkt, 0, bk, Kb, 0, D, D, 1.0f);
        gemm_wmma_nt<false, true, false><<<g, 256, 0, stream>>>(
            X, 0, Wvt, 0, bv, Vb, 0, D, D, 1.0f);
    }

    // 1b) V^T per batch for the attn@V NT GEMM
    transpose_bf16_batched<<<dim3(D / 32, S / 32, Bn), 256, 0, stream>>>(
        Vb, Vt, S, D, (long long)S * D, (long long)S * D);

    // 2) scores[b] = SCALE * Q_b @ K_b^T   (fp32 out, batched over B)
    {
        dim3 g(S / TILE_N, S / TILE_M, Bn);
        gemm_wmma_nt<false, false, true><<<g, 256, 0, stream>>>(
            Qb, (long long)S * D, Kb, (long long)S * D, nullptr,
            Sc, (long long)S * S, S, D, SCALE);
    }

    // 3) masked softmax rows -> bf16 attn
    softmax_mask<<<Bn * S, 256, 0, stream>>>(Sc, mask1, mask2, At, S, S / 2);

    // 4) attended[b] = attn_b @ V_b  ==  attn_b @ (Vt_b)^T   (bf16 out, batched)
    {
        dim3 g(D / TILE_N, S / TILE_M, Bn);
        gemm_wmma_nt<false, false, false><<<g, 256, 0, stream>>>(
            At, (long long)S * S, Vt, (long long)S * D, nullptr,
            Attd, (long long)S * D, D, S, 1.0f);
    }

    // 5) h1 = relu(attended @ W1 + b1)   (NT vs W1^T)
    {
        dim3 g(H1 / TILE_N, M / TILE_M, 1);
        gemm_wmma_nt<true, true, false><<<g, 256, 0, stream>>>(
            Attd, 0, W1t, 0, b1, h1, 0, H1, D, 1.0f);
    }

    // 6) h2 = relu(h1 @ W2 + b2)   (NT vs W2^T)
    {
        dim3 g(H2 / TILE_N, M / TILE_M, 1);
        gemm_wmma_nt<true, true, false><<<g, 256, 0, stream>>>(
            h1, 0, W2t, 0, b2, h2, 0, H2, H1, 1.0f);
    }

    // 7) logits -> sigmoid (N==1 reduction head)
    head_sigmoid<<<M, 256, 0, stream>>>(h2, W3, b3, (float*)d_out, H2);
}
